// AttentionLayer_36490042147490
// MI455X (gfx1250) — matrix operand order repaired and checked by reading, not compiled, not run on hardware
//
#include <hip/hip_runtime.h>

// Problem constants (match reference)
constexpr int Bc   = 2;
constexpr int Sc   = 2048;
constexpr int HIDc = 1024;
constexpr int Dc   = 64;
constexpr int Hc   = HIDc / Dc;   // 16
constexpr int ROTc = 32;
constexpr int Mc   = Bc * Sc;     // 4096 rows for the projections

typedef __attribute__((ext_vector_type(16))) __bf16 v16bf;
typedef __attribute__((ext_vector_type(8)))  float  v8f;

// ---------------- fragment loaders ----------------
// A fragment, 16x32 bf16, row-major source: lane holds row (lane%16),
// element j <-> K = (j<8 ? j : j+8) + 8*(lane/16)  => two 16B loads.
__device__ inline v16bf ldA(const __bf16* __restrict__ rowptr, int k0, int hf) {
    union { v16bf v; uint4 u[2]; } t;
    t.u[0] = *(const uint4*)(rowptr + k0 + 8 * hf);
    t.u[1] = *(const uint4*)(rowptr + k0 + 16 + 8 * hf);
    return t.v;
}
// B fragment, 32x16 bf16, column-contiguous source (Bt[n][k]): lane holds
// column (lane%16), element j <-> K = k0 + (j<8 ? j : j+8) + 8*(lane/16) => two 16B loads.
__device__ inline v16bf ldB(const __bf16* __restrict__ colptr, int k0, int hf) {
    union { v16bf v; uint4 u[2]; } t;
    const __bf16* p = colptr + k0 + 8 * hf;
    t.u[0] = *(const uint4*)(p);
    t.u[1] = *(const uint4*)(p + 16);
    return t.v;
}

#define WMMA_BF16(a, b, c) \
    __builtin_amdgcn_wmma_f32_16x16x32_bf16(false, (a), false, (b), (short)0, (c), false, false)

// ---------------- conversion kernels ----------------
__global__ void cvt_f32_bf16_kernel(const float* __restrict__ in, __bf16* __restrict__ out, int n) {
    int i = blockIdx.x * blockDim.x + threadIdx.x;
    if (i < n) out[i] = (__bf16)in[i];
}

// Wt[n*K + k] = (bf16) W[k*N + n]   (W is K x N row-major)
__global__ void transpose_cvt_kernel(const float* __restrict__ W, __bf16* __restrict__ Wt,
                                     int Ndim, int Kdim) {
    int i = blockIdx.x * blockDim.x + threadIdx.x;
    if (i < Ndim * Kdim) {
        int k = i % Kdim, n = i / Kdim;
        Wt[i] = (__bf16)W[(size_t)k * Ndim + n];
    }
}

// ---------------- generic bf16 WMMA GEMM ----------------
// C[M x N] = A[M x K] * Bt[N x K]^T + bias; out to bf16 (Cbf) or f32 (Cf).
// 8 waves / block; each wave owns a 32(M) x 64(N) tile:
//   2 A-fragments share 4 B-fragments per 32-K step -> 8 WMMA / 12 b128 loads.
__global__ __launch_bounds__(256) void gemm_bf16_kernel(
    const __bf16* __restrict__ A, const __bf16* __restrict__ Bt,
    const float* __restrict__ bias, __bf16* __restrict__ Cbf, float* __restrict__ Cf,
    int Mdim, int Ndim, int Kdim)
{
    const int lane = threadIdx.x & 31;
    const int wid  = threadIdx.x >> 5;
    const int r    = lane & 15;
    const int hf   = lane >> 4;

    const int tiles_n = Ndim >> 6;
    const int gt = blockIdx.x * 8 + wid;
    const int m0 = (gt / tiles_n) * 32;
    const int n0 = (gt % tiles_n) * 64;

    const __bf16* arow0 = A  + (size_t)(m0 + r) * Kdim;
    const __bf16* arow1 = A  + (size_t)(m0 + 16 + r) * Kdim;
    const __bf16* bc0   = Bt + (size_t)(n0 +  0 + r) * Kdim;
    const __bf16* bc1   = Bt + (size_t)(n0 + 16 + r) * Kdim;
    const __bf16* bc2   = Bt + (size_t)(n0 + 32 + r) * Kdim;
    const __bf16* bc3   = Bt + (size_t)(n0 + 48 + r) * Kdim;

    v8f a0 = {}, a1 = {}, a2 = {}, a3 = {};   // rows m0 .. m0+15
    v8f c0 = {}, c1 = {}, c2 = {}, c3 = {};   // rows m0+16 .. m0+31

    for (int k0 = 0; k0 < Kdim; k0 += 32) {
        __builtin_prefetch(arow0 + k0 + 128, 0, 0);   // global_prefetch_b8
        __builtin_prefetch(arow1 + k0 + 128, 0, 0);
        const v16bf af0 = ldA(arow0, k0, hf);
        const v16bf af1 = ldA(arow1, k0, hf);
        v16bf bf;
        bf = ldB(bc0, k0, hf);
        a0 = WMMA_BF16(af0, bf, a0);
        c0 = WMMA_BF16(af1, bf, c0);
        bf = ldB(bc1, k0, hf);
        a1 = WMMA_BF16(af0, bf, a1);
        c1 = WMMA_BF16(af1, bf, c1);
        bf = ldB(bc2, k0, hf);
        a2 = WMMA_BF16(af0, bf, a2);
        c2 = WMMA_BF16(af1, bf, c2);
        bf = ldB(bc3, k0, hf);
        a3 = WMMA_BF16(af0, bf, a3);
        c3 = WMMA_BF16(af1, bf, c3);
    }

    const float bi0 = bias ? bias[n0 +  0 + r] : 0.f;
    const float bi1 = bias ? bias[n0 + 16 + r] : 0.f;
    const float bi2 = bias ? bias[n0 + 32 + r] : 0.f;
    const float bi3 = bias ? bias[n0 + 48 + r] : 0.f;

#pragma unroll
    for (int i = 0; i < 8; i++) {
        const int rowA = m0 + i + 8 * hf;            // C layout: rows i + 8*half
        const int rowC = rowA + 16;
        const size_t baseA = (size_t)rowA * Ndim + n0;
        const size_t baseC = (size_t)rowC * Ndim + n0;
        if (Cf) {
            Cf[baseA +  0 + r] = a0[i] + bi0;
            Cf[baseA + 16 + r] = a1[i] + bi1;
            Cf[baseA + 32 + r] = a2[i] + bi2;
            Cf[baseA + 48 + r] = a3[i] + bi3;
            Cf[baseC +  0 + r] = c0[i] + bi0;
            Cf[baseC + 16 + r] = c1[i] + bi1;
            Cf[baseC + 32 + r] = c2[i] + bi2;
            Cf[baseC + 48 + r] = c3[i] + bi3;
        } else {
            Cbf[baseA +  0 + r] = (__bf16)(a0[i] + bi0);
            Cbf[baseA + 16 + r] = (__bf16)(a1[i] + bi1);
            Cbf[baseA + 32 + r] = (__bf16)(a2[i] + bi2);
            Cbf[baseA + 48 + r] = (__bf16)(a3[i] + bi3);
            Cbf[baseC +  0 + r] = (__bf16)(c0[i] + bi0);
            Cbf[baseC + 16 + r] = (__bf16)(c1[i] + bi1);
            Cbf[baseC + 32 + r] = (__bf16)(c2[i] + bi2);
            Cbf[baseC + 48 + r] = (__bf16)(c3[i] + bi3);
        }
    }
}

// ---------------- rotary + repack ----------------
// Reference rotary is elementwise: out[j] = x[j]*(cos[j] - sin[j]) for even j,
//                                  out[j] = x[j]*(cos[j] + sin[j]) for odd j  (j < ROT)
// q additionally scaled by 1/sqrt(D). Writes:
//   qr,kr : [b][h][s][d] bf16 (rows contiguous in d)
//   vt    : [b][h][d][s] bf16 (columns of V contiguous in s, for P*V B-fragments)
__global__ void rope_pack_kernel(const __bf16* __restrict__ qb, const __bf16* __restrict__ kb,
                                 const __bf16* __restrict__ vb, const float* __restrict__ sinu,
                                 __bf16* __restrict__ qr, __bf16* __restrict__ kr,
                                 __bf16* __restrict__ vt)
{
    int idx = blockIdx.x * blockDim.x + threadIdx.x;
    if (idx >= Bc * Sc * HIDc) return;
    int hid = idx % HIDc;
    int s   = (idx / HIDc) % Sc;
    int b   = idx / (HIDc * Sc);
    int h = hid / Dc, d = hid % Dc;

    float qv = (float)qb[idx];
    float kv = (float)kb[idx];
    float vv = (float)vb[idx];
    if (d < ROTc) {
        float sn = sinu[((size_t)(b * 2 + 0) * Sc + s) * ROTc + d];
        float cs = sinu[((size_t)(b * 2 + 1) * Sc + s) * ROTc + d];
        float f = (d & 1) ? (cs + sn) : (cs - sn);
        qv *= f; kv *= f;
    }
    qv *= 0.125f;  // 1/sqrt(64)

    size_t head = (size_t)b * Hc + h;
    qr[(head * Sc + s) * Dc + d] = (__bf16)qv;
    kr[(head * Sc + s) * Dc + d] = (__bf16)kv;
    vt[(head * Dc + d) * Sc + s] = (__bf16)vv;
}

// ---------------- flash attention ----------------
// One wave per (b, h, 16-query tile). Streams keys in chunks of 32.
// probs = softmax(scores) * bias  => acc += exp(s-m)*bias*v ; l += exp(s-m).
__global__ __launch_bounds__(32) void flash_attn_kernel(
    const __bf16* __restrict__ qr, const __bf16* __restrict__ kr,
    const __bf16* __restrict__ vt, const float* __restrict__ bias,
    __bf16* __restrict__ ao)
{
    __shared__ __bf16 pbuf[16 * 32];

    const int lane = threadIdx.x;
    const int r  = lane & 15;
    const int hf = lane >> 4;

    const int qt = blockIdx.x % (Sc / 16);
    const int h  = (blockIdx.x / (Sc / 16)) % Hc;
    const int b  =  blockIdx.x / ((Sc / 16) * Hc);
    const int q0 = qt * 16;

    const __bf16* qp = qr + (((size_t)b * Hc + h) * Sc + q0) * Dc;
    const __bf16* kp = kr + ((size_t)b * Hc + h) * Sc * Dc;
    const __bf16* vp = vt + ((size_t)b * Hc + h) * Dc * Sc;
    const float*  bp = bias + (size_t)b * Sc * Sc;

    const v16bf aq0 = ldA(qp + r * Dc, 0,  hf);   // q rows, K = d 0..31
    const v16bf aq1 = ldA(qp + r * Dc, 32, hf);   // q rows, K = d 32..63

    v8f acc0 = {}, acc1 = {}, acc2 = {}, acc3 = {};
    float mrow[8], lrow[8];
#pragma unroll
    for (int i = 0; i < 8; i++) { mrow[i] = -1e30f; lrow[i] = 0.f; }

    for (int key0 = 0; key0 < Sc; key0 += 32) {
        // scores: two 16x16 tiles, K-dim = D = 64 (two WMMA each)
        v8f s0 = {}, s1 = {};
        {
            const __bf16* kc = kp + (size_t)(key0 + r) * Dc;       // Kt columns contiguous
            s0 = WMMA_BF16(aq0, ldB(kc, 0, hf),  s0);
            s0 = WMMA_BF16(aq1, ldB(kc, 32, hf), s0);
        }
        {
            const __bf16* kc = kp + (size_t)(key0 + 16 + r) * Dc;
            s1 = WMMA_BF16(aq0, ldB(kc, 0, hf),  s1);
            s1 = WMMA_BF16(aq1, ldB(kc, 32, hf), s1);
        }

        // online softmax (row stats shared across the lane's 16-lane group)
#pragma unroll
        for (int i = 0; i < 8; i++) {
            float c = fmaxf(s0[i], s1[i]);
            c = fmaxf(c, __shfl_xor(c, 1));
            c = fmaxf(c, __shfl_xor(c, 2));
            c = fmaxf(c, __shfl_xor(c, 4));
            c = fmaxf(c, __shfl_xor(c, 8));
            const float mnew  = fmaxf(mrow[i], c);
            const float scale = __expf(mrow[i] - mnew);
            mrow[i] = mnew;
            acc0[i] *= scale; acc1[i] *= scale; acc2[i] *= scale; acc3[i] *= scale;
            const float e0 = __expf(s0[i] - mnew);
            const float e1 = __expf(s1[i] - mnew);
            lrow[i] = lrow[i] * scale + e0 + e1;   // per-lane partial; reduced at end
            const int row = q0 + i + 8 * hf;
            const float b0 = bp[(size_t)row * Sc + key0 + r];
            const float b1 = bp[(size_t)row * Sc + key0 + 16 + r];
            pbuf[(i + 8 * hf) * 32 + r]      = (__bf16)(e0 * b0);
            pbuf[(i + 8 * hf) * 32 + 16 + r] = (__bf16)(e1 * b1);
        }

        // relayout P (C-layout) -> A-fragment via LDS (per-wave DS is in-order)
        v16bf pf;
#pragma unroll
        for (int j = 0; j < 16; j++) {
            const int kk = ((j < 8) ? j : j + 8) + 8 * hf;
            pf[j] = pbuf[r * 32 + kk];
        }

        // P(16x32) * V(32x64): Vt columns (fixed d, contiguous s)
        acc0 = WMMA_BF16(pf, ldB(vp + (size_t)( 0 + r) * Sc, key0, hf), acc0);
        acc1 = WMMA_BF16(pf, ldB(vp + (size_t)(16 + r) * Sc, key0, hf), acc1);
        acc2 = WMMA_BF16(pf, ldB(vp + (size_t)(32 + r) * Sc, key0, hf), acc2);
        acc3 = WMMA_BF16(pf, ldB(vp + (size_t)(48 + r) * Sc, key0, hf), acc3);
    }

#pragma unroll
    for (int i = 0; i < 8; i++) {
        float l = lrow[i];
        l += __shfl_xor(l, 1);
        l += __shfl_xor(l, 2);
        l += __shfl_xor(l, 4);
        l += __shfl_xor(l, 8);
        const float inv = 1.0f / l;
        const int row = q0 + i + 8 * hf;
        const size_t base = ((size_t)b * Sc + row) * HIDc + h * Dc;
        ao[base +  0 + r] = (__bf16)(acc0[i] * inv);
        ao[base + 16 + r] = (__bf16)(acc1[i] * inv);
        ao[base + 32 + r] = (__bf16)(acc2[i] * inv);
        ao[base + 48 + r] = (__bf16)(acc3[i] * inv);
    }
}

// ---------------- host ----------------
extern "C" void kernel_launch(void* const* d_in, const int* in_sizes, int n_in,
                              void* d_out, int out_size, void* d_ws, size_t ws_size,
                              hipStream_t stream)
{
    const float* x    = (const float*)d_in[0];
    const float* sinu = (const float*)d_in[1];
    const float* ab   = (const float*)d_in[2];
    const float* Wq   = (const float*)d_in[3];
    const float* bq   = (const float*)d_in[4];
    const float* Wk   = (const float*)d_in[5];
    const float* bk   = (const float*)d_in[6];
    const float* Wv   = (const float*)d_in[7];
    const float* bv   = (const float*)d_in[8];
    const float* Wo   = (const float*)d_in[9];
    float* out = (float*)d_out;

    char* w = (char*)d_ws;
    size_t off = 0;
    auto nxt = [&](size_t elems) -> __bf16* {
        __bf16* p = (__bf16*)(w + off);
        off += ((elems * 2 + 255) & ~(size_t)255);
        return p;
    };

    const size_t NX = (size_t)Mc * HIDc;   // 4096*1024
    const size_t NW = (size_t)HIDc * HIDc; // 1024*1024

    __bf16* xb  = nxt(NX);
    __bf16* wqt = nxt(NW);
    __bf16* wkt = nxt(NW);
    __bf16* wvt = nxt(NW);
    __bf16* wot = nxt(NW);
    __bf16* qb  = nxt(NX);
    __bf16* kb  = nxt(NX);
    __bf16* vb  = nxt(NX);
    __bf16* qr  = nxt(NX);
    __bf16* krr = nxt(NX);
    __bf16* vtt = nxt(NX);
    __bf16* ao  = nxt(NX);

    const int thr = 256;
    const int gx  = (int)(NX / thr);            // 16384
    const int gw  = (int)(NW / thr);            // 4096
    const int gemm_blocks = (Mc / 32) * (HIDc / 64) / 8;  // 256

    cvt_f32_bf16_kernel<<<gx, thr, 0, stream>>>(x, xb, (int)NX);
    transpose_cvt_kernel<<<gw, thr, 0, stream>>>(Wq, wqt, HIDc, HIDc);
    transpose_cvt_kernel<<<gw, thr, 0, stream>>>(Wk, wkt, HIDc, HIDc);
    transpose_cvt_kernel<<<gw, thr, 0, stream>>>(Wv, wvt, HIDc, HIDc);
    transpose_cvt_kernel<<<gw, thr, 0, stream>>>(Wo, wot, HIDc, HIDc);

    gemm_bf16_kernel<<<gemm_blocks, thr, 0, stream>>>(xb, wqt, bq, qb, nullptr, Mc, HIDc, HIDc);
    gemm_bf16_kernel<<<gemm_blocks, thr, 0, stream>>>(xb, wkt, bk, kb, nullptr, Mc, HIDc, HIDc);
    gemm_bf16_kernel<<<gemm_blocks, thr, 0, stream>>>(xb, wvt, bv, vb, nullptr, Mc, HIDc, HIDc);

    rope_pack_kernel<<<gx, thr, 0, stream>>>(qb, kb, vb, sinu, qr, krr, vtt);

    flash_attn_kernel<<<Bc * Hc * (Sc / 16), 32, 0, stream>>>(qr, krr, vtt, ab, ao);

    gemm_bf16_kernel<<<gemm_blocks, thr, 0, stream>>>(ao, wot, nullptr, nullptr, out, Mc, HIDc, HIDc);
}
